// defomableConv_offset_13520557048343
// MI455X (gfx1250) — matrix operand order repaired and checked by reading, not compiled, not run on hardware
//
#include <hip/hip_runtime.h>
#include <stdint.h>

typedef __attribute__((ext_vector_type(16))) _Float16 v16h;
typedef __attribute__((ext_vector_type(8)))  float    v8f;

#define BB   8
#define CC   256
#define HWP  4096     // H*W
#define OG   2
#define CG   128      // C / OG
#define KN   9        // 3x3
#define OFFC 36       // OG*2*K
#define OUTC 256
#define OCG  128      // OUTC / OG
#define KC   1152     // CG * KN   (deform GEMM reduction)
#define NP1  36       // KC/32 panels (deform)
#define KC2  2304     // CC * KN   (offset GEMM reduction)
#define NP2  72       // KC2/32 panels (offset)
#define NOF  64       // offset GEMM N padded (36 -> 64, uniform async staging)

// ws layout (bytes)
#define WS_OFFS   0u
#define WS_WRE    4718592u                    //   589,824 B
#define WS_WRE2   5308416u                    //   294,912 B
#define WS_REGION 5603328u                    // im2col region (18,874,368 B / batch)
#define CHUNK_B   18874368u

static __device__ __forceinline__ uint32_t pack2(float a, float b) {
    union { _Float16 h[2]; uint32_t u; } r;
    r.h[0] = (_Float16)a; r.h[1] = (_Float16)b;
    return r.u;
}

// ---------------------------------------------------------------------------
// Weight pack (deform): wre[g][s][o][kk], kc = s*32+kk = k*CG + c
// ---------------------------------------------------------------------------
__global__ void reorder_w(const float* __restrict__ dw, _Float16* __restrict__ wre)
{
    int t  = blockIdx.x * blockDim.x + threadIdx.x;   // [0, OG*NP1*OCG*32)
    int kk = t & 31;
    int o  = (t >> 5) & 127;
    int t3 = t >> 12;
    int s  = t3 % NP1;
    int g  = t3 / NP1;
    int kc = s * 32 + kk;
    int k  = kc / CG, c = kc % CG;
    wre[t] = (_Float16)dw[((size_t)(g * OCG + o) * CG + c) * 9 + k];
}

// ---------------------------------------------------------------------------
// Weight pack (offset conv): wre2[s2][o(64)][kk], kc2 = s2*32+kk = k*CC + c
// o in [36,64) zero-padded.
// ---------------------------------------------------------------------------
__global__ void reorder_w2(const float* __restrict__ ow, _Float16* __restrict__ wre2)
{
    int t  = blockIdx.x * blockDim.x + threadIdx.x;   // [0, NP2*NOF*32)
    int kk = t & 31;
    int o  = (t >> 5) & 63;
    int s2 = t >> 11;
    int k  = s2 >> 3;
    int c  = ((s2 & 7) << 5) + kk;
    wre2[t] = (o < OFFC) ? (_Float16)ow[((size_t)o * CC + c) * 9 + k] : (_Float16)0.0f;
}

// ---------------------------------------------------------------------------
// im2col of feta3 (3x3, zero pad) -> f16 panels col2[s2][p][kk]; z = batch.
// ---------------------------------------------------------------------------
__global__ void im2col_feta(const float* __restrict__ feta3,
                            uint32_t* __restrict__ col2u_base,
                            int b0, size_t zstrideB)
{
    int b = b0 + blockIdx.z;
    uint32_t* col2u = (uint32_t*)((char*)col2u_base + blockIdx.z * zstrideB);
    int t = blockIdx.x * blockDim.x + threadIdx.x;    // [0, KN*HW) exact
    int p = t & (HWP - 1);
    int k = t >> 12;
    int h = p >> 6, w = p & 63;
    int ky = k / 3, kx = k % 3;
    int y = h + ky - 1, x = w + kx - 1;
    bool ok = ((unsigned)y < 64u) && ((unsigned)x < 64u);
    int lin = (ok ? y : 0) * 64 + (ok ? x : 0);

    const float* fb = feta3 + (size_t)b * CC * HWP;
    for (int c = 0; c < CC; c += 2) {
        float v0 = ok ? fb[(size_t)c * HWP + lin]       : 0.f;
        float v1 = ok ? fb[(size_t)(c + 1) * HWP + lin] : 0.f;
        int s2 = k * 8 + (c >> 5);
        col2u[((size_t)s2 * HWP + p) * 16 + ((c & 31) >> 1)] = pack2(v0, v1);
    }
}

// ---------------------------------------------------------------------------
// Bilinear sampling -> f16 panels col[g][s][p][kk]; y = group, z = batch.
// ---------------------------------------------------------------------------
__global__ void sample_im2col(const float* __restrict__ x,
                              const float* __restrict__ offs,
                              uint32_t* __restrict__ colu_base,
                              int b0, size_t zstrideB)
{
    int b = b0 + blockIdx.z;
    int g = blockIdx.y;
    uint32_t* colu = (uint32_t*)((char*)colu_base + blockIdx.z * zstrideB);
    int t = blockIdx.x * blockDim.x + threadIdx.x;    // [0, KN*HW) exact
    int p = t & (HWP - 1);
    int k = t >> 12;
    int h = p >> 6, w = p & 63;

    const float* ob = offs + ((size_t)b * OFFC + (size_t)(g * KN + k) * 2) * HWP;
    float dy = ob[p];
    float dx = ob[HWP + p];
    int ky = k / 3, kx = k % 3;
    float py = (float)(h + ky - 1) + dy;
    float px = (float)(w + kx - 1) + dx;
    float fy = floorf(py), fx = floorf(px);
    float wy = py - fy,    wx = px - fx;
    int y0 = (int)fy, x0 = (int)fx, y1 = y0 + 1, x1 = x0 + 1;

    float w00 = (1.f - wy) * (1.f - wx) * ((y0 >= 0 && y0 < 64 && x0 >= 0 && x0 < 64) ? 1.f : 0.f);
    float w01 = (1.f - wy) * wx         * ((y0 >= 0 && y0 < 64 && x1 >= 0 && x1 < 64) ? 1.f : 0.f);
    float w10 = wy * (1.f - wx)         * ((y1 >= 0 && y1 < 64 && x0 >= 0 && x0 < 64) ? 1.f : 0.f);
    float w11 = wy * wx                 * ((y1 >= 0 && y1 < 64 && x1 >= 0 && x1 < 64) ? 1.f : 0.f);

    int cy0 = min(max(y0, 0), 63), cx0 = min(max(x0, 0), 63);
    int cy1 = min(max(y1, 0), 63), cx1 = min(max(x1, 0), 63);
    int l00 = cy0 * 64 + cx0, l01 = cy0 * 64 + cx1;
    int l10 = cy1 * 64 + cx0, l11 = cy1 * 64 + cx1;

    const float* xb = x + ((size_t)b * CC + (size_t)g * CG) * HWP;
    uint32_t* cg = colu + (size_t)g * NP1 * HWP * 16;
    for (int c = 0; c < CG; c += 2) {
        const float* xc0 = xb + (size_t)c * HWP;
        const float* xc1 = xc0 + HWP;
        float v0 = w00 * xc0[l00] + w01 * xc0[l01] + w10 * xc0[l10] + w11 * xc0[l11];
        float v1 = w00 * xc1[l00] + w01 * xc1[l01] + w10 * xc1[l10] + w11 * xc1[l11];
        int s = k * 4 + (c >> 5);
        cg[((size_t)s * HWP + p) * 16 + ((c & 31) >> 1)] = pack2(v0, v1);
    }
}

// ---------------------------------------------------------------------------
// Fragment loaders (ISA 7.12.2 16-bit layouts), LDS tiles are [row][kk32].
// ---------------------------------------------------------------------------
static __device__ __forceinline__ v16h loadA(const uint32_t* lAu, int mrow, int lhalf)
{
    union { v16h h; uint32_t u[8]; } a;
    #pragma unroll
    for (int j = 0; j < 8; ++j) {
        int dw = (j < 4 ? j : j + 4) + lhalf * 4;
        a.u[j] = lAu[mrow * 16 + dw];
    }
    return a.h;
}
static __device__ __forceinline__ v16h loadB(const uint32_t* lBu, int o, int lhalf)
{
    union { v16h h; uint32_t u[8]; } bf;
    #pragma unroll
    for (int j = 0; j < 8; ++j)
        bf.u[j] = lBu[o * 16 + (j < 4 ? j : j + 4) + lhalf * 4];
    return bf.h;
}

// ---------------------------------------------------------------------------
// Offset-conv WMMA GEMM: M=4096, N=64(36), K=2304; z = batch.
// 128 threads / 4 waves, 64x64 block tile, double-buffered async staging.
// Epilogue fuses thermal-mask scaling.
// ---------------------------------------------------------------------------
__global__ __launch_bounds__(128) void offset_gemm_wmma(
    const _Float16* __restrict__ col2_base,
    const _Float16* __restrict__ wre2,
    const float* __restrict__ mask,
    const float* __restrict__ betap,
    float* __restrict__ offs,
    int b0, size_t zstrideB)
{
    __shared__ _Float16 lA[2][64 * 32];   // 2 x 4 KB
    __shared__ _Float16 lB[2][NOF * 32];  // 2 x 4 KB

    const int b = b0 + blockIdx.z;
    const _Float16* col2 = (const _Float16*)((const char*)col2_base + blockIdx.z * zstrideB);

    const int tid = threadIdx.x;
    const int lane = tid & 31, wv = tid >> 5;
    const int lhalf = (lane >= 16) ? 1 : 0;
    const int l15 = lane & 15;
    const int mBase = blockIdx.x * 64;
    const int mrow = wv * 16 + l15;

    const uint32_t off = tid * 16;
    uint32_t dA[2] = { (uint32_t)(uintptr_t)&lA[0][0] + off, (uint32_t)(uintptr_t)&lA[1][0] + off };
    uint32_t dB[2] = { (uint32_t)(uintptr_t)&lB[0][0] + off, (uint32_t)(uintptr_t)&lB[1][0] + off };

    v8f acc[3];
    #pragma unroll
    for (int i = 0; i < 3; ++i)
        #pragma unroll
        for (int e = 0; e < 8; ++e) acc[i][e] = 0.f;

    auto issue = [&](int s, int buf) {
        uint64_t gA = (uint64_t)(uintptr_t)(col2 + ((size_t)s * HWP + mBase) * 32);
        uint64_t gB = (uint64_t)(uintptr_t)(wre2 + (size_t)s * NOF * 32);
        asm volatile(
            "global_load_async_to_lds_b128 %0, %2, %3\n\t"
            "global_load_async_to_lds_b128 %0, %2, %3 offset:2048\n\t"
            "global_load_async_to_lds_b128 %1, %2, %4\n\t"
            "global_load_async_to_lds_b128 %1, %2, %4 offset:2048"
            :: "v"(dA[buf]), "v"(dB[buf]), "v"(off), "s"(gA), "s"(gB) : "memory");
    };
    auto compute = [&](int buf) {
        const uint32_t* lAu = (const uint32_t*)lA[buf];
        const uint32_t* lBu = (const uint32_t*)lB[buf];
        v16h a = loadA(lAu, mrow, lhalf);
        #pragma unroll
        for (int nt = 0; nt < 3; ++nt) {
            v16h bf = loadB(lBu, nt * 16 + l15, lhalf);
            acc[nt] = __builtin_amdgcn_wmma_f32_16x16x32_f16(
                false, a, false, bf, (short)0, acc[nt], false, false);
        }
    };

    issue(0, 0);
    for (int s = 0; s < NP2 - 1; ++s) {
        int cur = s & 1;
        issue(s + 1, cur ^ 1);
        asm volatile("s_wait_asynccnt 0x4" ::: "memory");  // batch s landed
        __syncthreads();
        compute(cur);
        __syncthreads();
    }
    asm volatile("s_wait_asynccnt 0x0" ::: "memory");
    __syncthreads();
    compute((NP2 - 1) & 1);

    float beta = betap[0];
    #pragma unroll
    for (int nt = 0; nt < 3; ++nt) {
        int co = nt * 16 + l15;
        if (co < OFFC) {
            float* ob = offs + ((size_t)b * OFFC + co) * HWP;
            #pragma unroll
            for (int r = 0; r < 8; ++r) {
                int p = mBase + wv * 16 + r + lhalf * 8;
                float msc = 1.0f + beta * mask[(size_t)b * HWP + p];
                ob[p] = acc[nt][r] * msc;
            }
        }
    }
}

// ---------------------------------------------------------------------------
// Deform WMMA GEMM: M=4096, N=128, K=1152; y = group, z = batch.
// 256 threads / 8 waves, 128x128 block tile, double-buffered async staging.
// ---------------------------------------------------------------------------
__global__ __launch_bounds__(256) void deform_gemm_wmma(
    const _Float16* __restrict__ col_base,
    const _Float16* __restrict__ wre,
    const float* __restrict__ bias,
    float* __restrict__ out,
    int b0, size_t zstrideB)
{
    __shared__ _Float16 lA[2][128 * 32];  // 2 x 8 KB
    __shared__ _Float16 lB[2][128 * 32];  // 2 x 8 KB

    const int b = b0 + blockIdx.z;
    const int g = blockIdx.y;
    const _Float16* col = (const _Float16*)((const char*)col_base + blockIdx.z * zstrideB);
    const _Float16* colg = col + (size_t)g * NP1 * HWP * 32;
    const _Float16* wg   = wre + (size_t)g * NP1 * OCG * 32;

    const int tid = threadIdx.x;
    const int lane = tid & 31, wv = tid >> 5;
    const int lhalf = (lane >= 16) ? 1 : 0;
    const int l15 = lane & 15;
    const int mBase = blockIdx.x * 128;
    const int mrow = wv * 16 + l15;

    const uint32_t off = tid * 16;
    uint32_t dA[2] = { (uint32_t)(uintptr_t)&lA[0][0] + off, (uint32_t)(uintptr_t)&lA[1][0] + off };
    uint32_t dB[2] = { (uint32_t)(uintptr_t)&lB[0][0] + off, (uint32_t)(uintptr_t)&lB[1][0] + off };

    v8f acc[8];
    #pragma unroll
    for (int i = 0; i < 8; ++i)
        #pragma unroll
        for (int e = 0; e < 8; ++e) acc[i][e] = 0.f;

    auto issue = [&](int s, int buf) {
        uint64_t gA = (uint64_t)(uintptr_t)(colg + ((size_t)s * HWP + mBase) * 32);
        uint64_t gB = (uint64_t)(uintptr_t)(wg + (size_t)s * OCG * 32);
        asm volatile(
            "global_load_async_to_lds_b128 %0, %2, %3\n\t"
            "global_load_async_to_lds_b128 %0, %2, %3 offset:4096\n\t"
            "global_load_async_to_lds_b128 %1, %2, %4\n\t"
            "global_load_async_to_lds_b128 %1, %2, %4 offset:4096"
            :: "v"(dA[buf]), "v"(dB[buf]), "v"(off), "s"(gA), "s"(gB) : "memory");
    };
    auto compute = [&](int buf) {
        const uint32_t* lAu = (const uint32_t*)lA[buf];
        const uint32_t* lBu = (const uint32_t*)lB[buf];
        v16h a = loadA(lAu, mrow, lhalf);
        #pragma unroll
        for (int nt = 0; nt < 8; ++nt) {
            v16h bf = loadB(lBu, nt * 16 + l15, lhalf);
            acc[nt] = __builtin_amdgcn_wmma_f32_16x16x32_f16(
                false, a, false, bf, (short)0, acc[nt], false, false);
        }
    };

    issue(0, 0);
    for (int s = 0; s < NP1 - 1; ++s) {
        int cur = s & 1;
        issue(s + 1, cur ^ 1);
        asm volatile("s_wait_asynccnt 0x4" ::: "memory");  // batch s landed
        __syncthreads();
        compute(cur);
        __syncthreads();
    }
    asm volatile("s_wait_asynccnt 0x0" ::: "memory");
    __syncthreads();
    compute((NP1 - 1) & 1);

    #pragma unroll
    for (int nt = 0; nt < 8; ++nt) {
        int oc = g * OCG + nt * 16 + l15;
        float bv = bias[oc];
        float* ob = out + ((size_t)b * OUTC + oc) * HWP;
        #pragma unroll
        for (int r = 0; r < 8; ++r) {
            int p = mBase + wv * 16 + r + lhalf * 8;
            ob[p] = acc[nt][r] + bv;
        }
    }
}

// ---------------------------------------------------------------------------
extern "C" void kernel_launch(void* const* d_in, const int* in_sizes, int n_in,
                              void* d_out, int out_size, void* d_ws, size_t ws_size,
                              hipStream_t stream)
{
    (void)in_sizes; (void)n_in; (void)out_size;
    const float* feta3 = (const float*)d_in[0];
    const float* x     = (const float*)d_in[1];
    const float* mask  = (const float*)d_in[2];
    const float* beta  = (const float*)d_in[3];
    const float* ow    = (const float*)d_in[4];
    const float* dw    = (const float*)d_in[5];
    const float* db    = (const float*)d_in[6];
    float* out = (float*)d_out;

    char* ws = (char*)d_ws;
    float*    offs   = (float*)(ws + WS_OFFS);
    _Float16* wre    = (_Float16*)(ws + WS_WRE);
    _Float16* wre2   = (_Float16*)(ws + WS_WRE2);
    char*     region = ws + WS_REGION;   // im2col chunks (offset & deform stages share)

    const bool full = ws_size >= (size_t)WS_REGION + (size_t)BB * CHUNK_B;
    const int    nz = full ? BB : 1;
    const size_t zs = full ? (size_t)CHUNK_B : 0;

    reorder_w <<<(OG * NP1 * OCG * 32) / 256, 256, 0, stream>>>(dw, wre);
    reorder_w2<<<(NP2 * NOF * 32)      / 256, 256, 0, stream>>>(ow, wre2);

    if (full) {
        im2col_feta     <<<dim3(144, 1, BB), 256, 0, stream>>>(feta3, (uint32_t*)region, 0, zs);
        offset_gemm_wmma<<<dim3(64, 1, BB), 128, 0, stream>>>((const _Float16*)region, wre2, mask, beta, offs, 0, zs);
        sample_im2col   <<<dim3(144, 2, BB), 256, 0, stream>>>(x, offs, (uint32_t*)region, 0, zs);
        deform_gemm_wmma<<<dim3(32, 2, BB), 256, 0, stream>>>((const _Float16*)region, wre, db, out, 0, zs);
    } else {
        for (int b = 0; b < BB; ++b) {
            im2col_feta     <<<dim3(144, 1, 1), 256, 0, stream>>>(feta3, (uint32_t*)region, b, 0);
            offset_gemm_wmma<<<dim3(64, 1, 1), 128, 0, stream>>>((const _Float16*)region, wre2, mask, beta, offs, b, 0);
        }
        for (int b = 0; b < BB; ++b) {
            sample_im2col   <<<dim3(144, 2, 1), 256, 0, stream>>>(x, offs, (uint32_t*)region, b, 0);
            deform_gemm_wmma<<<dim3(32, 2, 1), 256, 0, stream>>>((const _Float16*)region, wre, db, out, b, 0);
        }
    }
    (void)nz;
}
